// MinLSTMLayer_8160437862911
// MI455X (gfx1250) — compile-verified, matrix operand order checked by reading, not run on hardware
//
#include <hip/hip_runtime.h>
#include <hip/hip_bf16.h>
#include <math.h>

// ---------------------------------------------------------------------------
// MinLSTM fused pipeline on gfx1250:
//   0) fp32 -> bf16 conversion pass for x and the 3 weight matrices
//   1) 3-gate NT GEMM with v_wmma_f32_16x16x32_bf16, double-buffered LDS fed
//      by GLOBAL_LOAD_ASYNC_TO_LDS_B128 (ASYNCcnt), fused gate math -> (f, v)
//   2-4) chunked linear scan  h_t = f_t*h_{t-1} + v_t,  h_0 = 1
// ---------------------------------------------------------------------------

typedef __attribute__((ext_vector_type(16))) __bf16     v16bf;
typedef __attribute__((ext_vector_type(8)))  float      v8f;
typedef __attribute__((ext_vector_type(4)))  unsigned   v4u;

#define BATCH 4
#define TQ    4096
#define DIN   1024
#define DH    1024
#define MTOT  (BATCH * TQ)      // 16384 rows

#define BM 128                  // block M tile
#define BN 64                   // block N tile
#define BK 64                   // staged K tile (2 WMMA K-steps of 32)
#define LDR 72                  // padded LDS row stride in bf16 (144 B)
#define KTILES (DIN / BK)       // 16

#define NCHUNK 64               // scan chunks along T
#define LCHUNK 64               // T / NCHUNK

// round-to-nearest-even fp32 -> bf16 bits
__device__ __forceinline__ unsigned short f2bf(float f) {
    unsigned u = __float_as_uint(f);
    u += 0x7FFFu + ((u >> 16) & 1u);
    return (unsigned short)(u >> 16);
}

// stable softplus
__device__ __forceinline__ float sp(float x) {
    return fmaxf(x, 0.f) + log1pf(__expf(-fabsf(x)));
}

union LdsFrag { v4u u[2]; v16bf v; };
__device__ __forceinline__ v16bf load_frag(const unsigned short* p) {
    LdsFrag f;
    f.u[0] = *reinterpret_cast<const v4u*>(p);       // 16B aligned
    f.u[1] = *reinterpret_cast<const v4u*>(p + 8);
    return f.v;
}

// CDNA5 async DMA: copy 16 bytes global -> LDS, tracked by ASYNCcnt.
__device__ __forceinline__ void async_b128(unsigned ldsOff, const void* g) {
    asm volatile("global_load_async_to_lds_b128 %0, %1, off"
                 :: "v"(ldsOff), "v"((unsigned long long)(uintptr_t)g)
                 : "memory");
}
__device__ __forceinline__ void wait_async_le(void) {   // <= 10 outstanding
    asm volatile("s_wait_asynccnt 0xa" ::: "memory");
}
__device__ __forceinline__ void wait_async_0(void) {
    asm volatile("s_wait_asynccnt 0x0" ::: "memory");
}

// ---------------------------------------------------------------------------
// Kernel 0: fp32 -> bf16 pack (8 elements / thread, 16B stores)
// ---------------------------------------------------------------------------
__global__ __launch_bounds__(256)
void cvt_f32_bf16(const float* __restrict__ src, unsigned short* __restrict__ dst,
                  int n)
{
    int i = (blockIdx.x * 256 + threadIdx.x) * 8;
    if (i >= n) return;
    float4 a = *reinterpret_cast<const float4*>(src + i);
    float4 b = *reinterpret_cast<const float4*>(src + i + 4);
    v4u o;
    o.x = (unsigned)f2bf(a.x) | ((unsigned)f2bf(a.y) << 16);
    o.y = (unsigned)f2bf(a.z) | ((unsigned)f2bf(a.w) << 16);
    o.z = (unsigned)f2bf(b.x) | ((unsigned)f2bf(b.y) << 16);
    o.w = (unsigned)f2bf(b.z) | ((unsigned)f2bf(b.w) << 16);
    *reinterpret_cast<v4u*>(dst + i) = o;
}

// ---------------------------------------------------------------------------
// Kernel 1: fused 3-gate GEMM.  grid = (M/128, N/64), 256 threads (8 waves).
// Double-buffered LDS, async global->LDS staging, 24 WMMAs per staged tile.
// ---------------------------------------------------------------------------
__global__ __launch_bounds__(256)
void minlstm_gates_wmma(const unsigned short* __restrict__ xb,
                        const unsigned short* __restrict__ Wfb,
                        const unsigned short* __restrict__ Wib,
                        const unsigned short* __restrict__ Whb,
                        const float* __restrict__ bff,
                        const float* __restrict__ bii,
                        const float* __restrict__ bhh,
                        float* __restrict__ F, float* __restrict__ V)
{
    __shared__ unsigned short As[2][BM][LDR];        // 2 x 18 KB
    __shared__ unsigned short Bs[2][3][BN][LDR];     // 2 x 27 KB

    const int t  = threadIdx.x;
    const int m0 = blockIdx.x * BM;
    const int n0 = blockIdx.y * BN;

    const int w  = t >> 5;          // wave 0..7
    const int l  = t & 31;
    const int wm = w & 3;           // 4 M-subtiles of 32
    const int wn = w >> 2;          // 2 N-subtiles of 32
    const int lm = l & 15;
    const int lk = (l >> 4) << 4;   // 0 or 16

    const unsigned short* Wg[3] = { Wfb, Wib, Whb };

    const unsigned asBase = (unsigned)(uintptr_t)&As[0][0][0];
    const unsigned bsBase = (unsigned)(uintptr_t)&Bs[0][0][0][0];
    const unsigned A_BUF  = BM * LDR * 2;            // bytes per A buffer
    const unsigned B_BUF  = 3 * BN * LDR * 2;        // bytes per B buffer (3 gates)
    const unsigned B_GATE = BN * LDR * 2;

    // async staging assignment: 16B chunks, 8 chunks per row
    const int aRow0 = t >> 3, aCol = t & 7;          // A: rows t>>3 + 32*q
    const int bRow0 = t >> 3, bCol = t & 7;          // B: rows t>>3 + 32*q (q=0,1)

    auto issue_tile = [&](int kk, int buf) {
        // A tile: 128 rows x 8 chunks = 1024 chunks, 4 per thread
#pragma unroll
        for (int q = 0; q < 4; ++q) {
            int row = aRow0 + q * 32;
            async_b128(asBase + (unsigned)buf * A_BUF + row * (LDR * 2) + aCol * 16,
                       xb + (size_t)(m0 + row) * DIN + kk + aCol * 8);
        }
        // B tiles: 3 gates x 64 rows x 8 chunks, 6 per thread
#pragma unroll
        for (int g = 0; g < 3; ++g) {
#pragma unroll
            for (int q = 0; q < 2; ++q) {
                int row = bRow0 + q * 32;
                async_b128(bsBase + (unsigned)buf * B_BUF + g * B_GATE +
                               row * (LDR * 2) + bCol * 16,
                           Wg[g] + (size_t)(n0 + row) * DIN + kk + bCol * 8);
            }
        }
    };

    v8f acc[3][2][2];
#pragma unroll
    for (int g = 0; g < 3; ++g)
#pragma unroll
        for (int i = 0; i < 2; ++i)
#pragma unroll
            for (int j = 0; j < 2; ++j)
                acc[g][i][j] = (v8f){0.f,0.f,0.f,0.f,0.f,0.f,0.f,0.f};

    issue_tile(0, 0);

    for (int it = 0; it < KTILES; ++it) {
        const int buf = it & 1;
        if (it + 1 < KTILES) {
            issue_tile((it + 1) * BK, (it + 1) & 1);
            wait_async_le();                 // tile `it` (first 10 in order) done
        } else {
            wait_async_0();
        }
        __syncthreads();                     // tile `it` visible to all waves

#pragma unroll
        for (int s = 0; s < 2; ++s) {        // two K=32 sub-steps
            const int ko = s * 32 + lk;
            v16bf a0 = load_frag(&As[buf][wm * 32 +  0 + lm][ko]);
            v16bf a1 = load_frag(&As[buf][wm * 32 + 16 + lm][ko]);
#pragma unroll
            for (int g = 0; g < 3; ++g) {
                v16bf b0 = load_frag(&Bs[buf][g][wn * 32 +  0 + lm][ko]);
                v16bf b1 = load_frag(&Bs[buf][g][wn * 32 + 16 + lm][ko]);
                acc[g][0][0] = __builtin_amdgcn_wmma_f32_16x16x32_bf16(
                    false, a0, false, b0, (short)0, acc[g][0][0], false, false);
                acc[g][0][1] = __builtin_amdgcn_wmma_f32_16x16x32_bf16(
                    false, a0, false, b1, (short)0, acc[g][0][1], false, false);
                acc[g][1][0] = __builtin_amdgcn_wmma_f32_16x16x32_bf16(
                    false, a1, false, b0, (short)0, acc[g][1][0], false, false);
                acc[g][1][1] = __builtin_amdgcn_wmma_f32_16x16x32_bf16(
                    false, a1, false, b1, (short)0, acc[g][1][1], false, false);
            }
        }
        __syncthreads();   // all waves done reading buf before it is re-filled
    }

    // ---- epilogue: bias + gate math, store f = exp(log_f), v = exp(log_i+log_g)
    const int mBase = m0 + wm * 32 + ((l >> 4) << 3);
    const int nBase = n0 + wn * 32 + lm;
#pragma unroll
    for (int j = 0; j < 2; ++j) {
        const int n = nBase + j * 16;
        const float bfv = bff[n], biv = bii[n], bhv = bhh[n];
#pragma unroll
        for (int i = 0; i < 2; ++i) {
#pragma unroll
            for (int r = 0; r < 8; ++r) {
                const int m = mBase + i * 16 + r;
                float fp = acc[0][i][j][r] + bfv;
                float ip = acc[1][i][j][r] + biv;
                float hp = acc[2][i][j][r] + bhv;
                float diff = sp(-fp) - sp(-ip);
                float logf = -sp(diff);
                float logi = -sp(-diff);
                float logg = (hp >= 0.f) ? __logf(hp + 0.5f) : -sp(-hp);
                size_t idx = (size_t)m * DH + n;
                F[idx] = __expf(logf);
                V[idx] = __expf(logi + logg);
            }
        }
    }
}

// ---------------------------------------------------------------------------
// Kernel 2: per-chunk scan summaries. One thread per (b, chunk, d).
// ---------------------------------------------------------------------------
__global__ __launch_bounds__(256)
void scan_pass1(const float* __restrict__ F, const float* __restrict__ V,
                float* __restrict__ cA, float* __restrict__ cH)
{
    const int tid = blockIdx.x * 256 + threadIdx.x;      // B*NCHUNK*DH
    const int d = tid & (DH - 1);
    const int c = (tid / DH) & (NCHUNK - 1);
    const int b = tid / (DH * NCHUNK);
    size_t base = ((size_t)(b * TQ + c * LCHUNK)) * DH + d;
    float A = 1.f, H = 0.f;
#pragma unroll 4
    for (int s = 0; s < LCHUNK; ++s) {
        float f = F[base + (size_t)s * DH];
        float v = V[base + (size_t)s * DH];
        A *= f;
        H = fmaf(f, H, v);
    }
    cA[tid] = A;
    cH[tid] = H;
}

// ---------------------------------------------------------------------------
// Kernel 3: sequential combine across chunks. One thread per (b, d). h_0 = 1.
// ---------------------------------------------------------------------------
__global__ __launch_bounds__(256)
void scan_combine(const float* __restrict__ cA, const float* __restrict__ cH,
                  float* __restrict__ cS)
{
    const int tid = blockIdx.x * 256 + threadIdx.x;      // B*DH
    const int d = tid & (DH - 1);
    const int b = tid / DH;
    float h = 1.f;
    for (int c = 0; c < NCHUNK; ++c) {
        size_t idx = ((size_t)(b * NCHUNK + c)) * DH + d;
        cS[idx] = h;
        h = fmaf(cA[idx], h, cH[idx]);
    }
}

// ---------------------------------------------------------------------------
// Kernel 4: final scan with chunk-entry state, write outputs.
// ---------------------------------------------------------------------------
__global__ __launch_bounds__(256)
void scan_pass2(const float* __restrict__ F, const float* __restrict__ V,
                const float* __restrict__ cS, float* __restrict__ out)
{
    const int tid = blockIdx.x * 256 + threadIdx.x;      // B*NCHUNK*DH
    const int d = tid & (DH - 1);
    const int c = (tid / DH) & (NCHUNK - 1);
    const int b = tid / (DH * NCHUNK);
    size_t base = ((size_t)(b * TQ + c * LCHUNK)) * DH + d;
    float h = cS[tid];
#pragma unroll 4
    for (int s = 0; s < LCHUNK; ++s) {
        size_t idx = base + (size_t)s * DH;
        h = fmaf(F[idx], h, V[idx]);
        out[idx] = h;
    }
}

// ---------------------------------------------------------------------------
extern "C" void kernel_launch(void* const* d_in, const int* in_sizes, int n_in,
                              void* d_out, int out_size, void* d_ws, size_t ws_size,
                              hipStream_t stream)
{
    const float* x  = (const float*)d_in[0];
    const float* Wf = (const float*)d_in[1];
    const float* bf = (const float*)d_in[2];
    const float* Wi = (const float*)d_in[3];
    const float* bi = (const float*)d_in[4];
    const float* Wh = (const float*)d_in[5];
    const float* bh = (const float*)d_in[6];
    float* out = (float*)d_out;

    // workspace layout (bytes)
    const size_t FV   = (size_t)MTOT * DH * sizeof(float);            // 64 MB
    const size_t XBF  = (size_t)MTOT * DIN * sizeof(unsigned short);  // 32 MB
    const size_t WBF  = (size_t)DH * DIN * sizeof(unsigned short);    //  2 MB
    const size_t CHNK = (size_t)BATCH * NCHUNK * DH * sizeof(float);  //  1 MB
    char* ws = (char*)d_ws;
    float*          F   = (float*)(ws);
    float*          V   = (float*)(ws + FV);
    unsigned short* xbf = (unsigned short*)(ws + 2 * FV);
    unsigned short* wfb = (unsigned short*)(ws + 2 * FV + XBF);
    unsigned short* wib = (unsigned short*)(ws + 2 * FV + XBF + WBF);
    unsigned short* whb = (unsigned short*)(ws + 2 * FV + XBF + 2 * WBF);
    float*          cA  = (float*)(ws + 2 * FV + XBF + 3 * WBF);
    float*          cH  = (float*)(ws + 2 * FV + XBF + 3 * WBF + CHNK);
    float*          cS  = (float*)(ws + 2 * FV + XBF + 3 * WBF + 2 * CHNK);

    // 0) fp32 -> bf16
    const int nX = MTOT * DIN;              // 16,777,216
    const int nW = DH * DIN;                //  1,048,576
    cvt_f32_bf16<<<nX / (256 * 8), 256, 0, stream>>>(x,  xbf, nX);
    cvt_f32_bf16<<<nW / (256 * 8), 256, 0, stream>>>(Wf, wfb, nW);
    cvt_f32_bf16<<<nW / (256 * 8), 256, 0, stream>>>(Wi, wib, nW);
    cvt_f32_bf16<<<nW / (256 * 8), 256, 0, stream>>>(Wh, whb, nW);

    // 1) fused WMMA GEMM + gate math
    dim3 ggrid(MTOT / BM, DH / BN);                                   // 128 x 16
    minlstm_gates_wmma<<<ggrid, 256, 0, stream>>>(xbf, wfb, wib, whb,
                                                  bf, bi, bh, F, V);

    // 2-4) chunked scan
    const int nChunkThreads = BATCH * NCHUNK * DH;                    // 262144
    scan_pass1<<<nChunkThreads / 256, 256, 0, stream>>>(F, V, cA, cH);
    const int nChanThreads = BATCH * DH;                              // 4096
    scan_combine<<<nChanThreads / 256, 256, 0, stream>>>(cA, cH, cS);
    scan_pass2<<<nChunkThreads / 256, 256, 0, stream>>>(F, V, cS, out);
}
